// RNNBlock_21045339750884
// MI455X (gfx1250) — compile-verified, matrix operand order checked by reading, not run on hardware
//
#include <hip/hip_runtime.h>

#define Bc 2
#define Sc 2048
#define Dc 1024
#define Hc 8
#define DKc 128
#define DVc 128
#define FFc 2816
#define Mc (Bc*Sc)   // 4096 rows

typedef __attribute__((ext_vector_type(16))) __bf16 v16bf;
typedef __attribute__((ext_vector_type(8)))  __bf16 v8bf;
typedef __attribute__((ext_vector_type(8)))  float  v8f;

union V16U { v16bf v; v8bf h[2]; };

// ---------------------------------------------------------------------------
// GEMM: C[M,N] (f32) = A[M,K] (bf16 row-major) * Bw[N,K]^T (bf16 row-major)
// Optional epilogue: C += Res. Block tile 128x128, K-step 32, 8 waves (4x2),
// each wave computes 32x64 via 2x4 v_wmma_f32_16x16x32_bf16 tiles.
// Global->LDS staging uses CDNA5 async loads (GLOBAL_LOAD_ASYNC_TO_LDS_B128,
// ASYNCcnt) with double-buffered LDS: one barrier per K-step, async fill of
// the next tile overlaps WMMA on the current tile.
// ---------------------------------------------------------------------------
template<int ADD_RES>
__global__ __launch_bounds__(256) void gemm_bf16(const __bf16* __restrict__ A,
                                                 const __bf16* __restrict__ Bw,
                                                 float* __restrict__ C,
                                                 const float* __restrict__ Res,
                                                 int M, int N, int K)
{
  constexpr int BM = 128, BN = 128, BK = 32, LDT = 48;  // 96B LDS row pitch
  constexpr int TILE = BM * LDT;                        // elems per buffer
  __shared__ __bf16 As[2 * TILE];
  __shared__ __bf16 Bs[2 * TILE];

  const int tid  = threadIdx.x;
  const int lane = tid & 31;
  const int wave = tid >> 5;
  const int wm   = wave >> 1;   // 0..3
  const int wn   = wave & 1;    // 0..1
  const int bm   = blockIdx.y * BM;
  const int bn   = blockIdx.x * BN;

  v8f acc[2][4];
  #pragma unroll
  for (int r = 0; r < 2; ++r)
    #pragma unroll
    for (int f = 0; f < 4; ++f)
      #pragma unroll
      for (int g = 0; g < 8; ++g)
        acc[r][f][g] = 0.0f;

  const int lrow = tid >> 1;          // 0..127: staging row
  const int lcol = (tid & 1) * 16;    // 0 or 16
  const int l15  = lane & 15;
  const int kb0  = (lane >> 4) * 8;   // ISA A/B layout: lanes0-15 K{0-7,16-23}, 16-31 K{8-15,24-31}

  // per-thread staging addresses (32B per thread per tile, as 2x b128 async)
  const __bf16* gA = A  + (size_t)(bm + lrow) * K + lcol;
  const __bf16* gB = Bw + (size_t)(bn + lrow) * K + lcol;
  const unsigned ldsA0 = (unsigned)(size_t)(As + lrow * LDT + lcol);
  const unsigned ldsB0 = (unsigned)(size_t)(Bs + lrow * LDT + lcol);

  auto issue_tile = [&](int buf, int k0) {
#if defined(__HIP_DEVICE_COMPILE__)
    const unsigned la = ldsA0 + (unsigned)(buf * TILE * 2);
    const unsigned lb = ldsB0 + (unsigned)(buf * TILE * 2);
    const __bf16* pa = gA + k0;
    const __bf16* pb = gB + k0;
    asm volatile(
      "global_load_async_to_lds_b128 %0, %2, off\n\t"
      "global_load_async_to_lds_b128 %0, %2, off offset:16\n\t"
      "global_load_async_to_lds_b128 %1, %3, off\n\t"
      "global_load_async_to_lds_b128 %1, %3, off offset:16"
      :: "v"(la), "v"(lb), "v"(pa), "v"(pb) : "memory");
#else
    (void)buf; (void)k0;
#endif
  };

  issue_tile(0, 0);
  int buf = 0;
  for (int k0 = 0; k0 < K; k0 += BK) {
#if defined(__HIP_DEVICE_COMPILE__)
    asm volatile("s_wait_asynccnt 0x0" ::: "memory");
#endif
    __syncthreads();                    // all waves' async fills of `buf` visible
    if (k0 + BK < K) issue_tile(buf ^ 1, k0 + BK);  // overlap next fill with WMMA

    const __bf16* Ab = As + buf * TILE;
    const __bf16* Bb = Bs + buf * TILE;

    V16U af[2];
    #pragma unroll
    for (int r = 0; r < 2; ++r) {
      const int arow = wm * 32 + r * 16 + l15;
      af[r].h[0] = *reinterpret_cast<const v8bf*>(Ab + arow * LDT + kb0);
      af[r].h[1] = *reinterpret_cast<const v8bf*>(Ab + arow * LDT + kb0 + 16);
    }
    #pragma unroll
    for (int f = 0; f < 4; ++f) {
      const int brow = wn * 64 + f * 16 + l15;
      V16U bfv;
      bfv.h[0] = *reinterpret_cast<const v8bf*>(Bb + brow * LDT + kb0);
      bfv.h[1] = *reinterpret_cast<const v8bf*>(Bb + brow * LDT + kb0 + 16);
#if defined(__HIP_DEVICE_COMPILE__)
      acc[0][f] = __builtin_amdgcn_wmma_f32_16x16x32_bf16(false, af[0].v, false, bfv.v,
                                                          (short)0, acc[0][f], false, false);
      acc[1][f] = __builtin_amdgcn_wmma_f32_16x16x32_bf16(false, af[1].v, false, bfv.v,
                                                          (short)0, acc[1][f], false, false);
#endif
    }
    buf ^= 1;
  }

  // epilogue: C/D layout: VGPR g -> row g (lanes 0-15) / g+8 (lanes 16-31), col = lane&15
  #pragma unroll
  for (int r = 0; r < 2; ++r)
    #pragma unroll
    for (int f = 0; f < 4; ++f)
      #pragma unroll
      for (int g = 0; g < 8; ++g) {
        const int row = bm + wm * 32 + r * 16 + g + 8 * (lane >> 4);
        const int col = bn + wn * 64 + f * 16 + l15;
        const size_t idx = (size_t)row * N + col;
        float vv = acc[r][f][g];
        if (ADD_RES) vv += Res[idx];
        C[idx] = vv;
      }
}

// ---------------------------------------------------------------------------
// RMSNorm over rows of length D=1024 -> bf16
// ---------------------------------------------------------------------------
__global__ __launch_bounds__(256) void rmsnorm_k(const float* __restrict__ x,
                                                 const float* __restrict__ w,
                                                 __bf16* __restrict__ out)
{
  __shared__ float red[8];
  __shared__ float s_scale;
  const int row = blockIdx.x;
  const int tid = threadIdx.x;
  const float* xr = x + (size_t)row * Dc;
  float v[4];
  float ss = 0.f;
  #pragma unroll
  for (int i = 0; i < 4; ++i) { v[i] = xr[tid + i * 256]; ss += v[i] * v[i]; }
  #pragma unroll
  for (int off = 16; off >= 1; off >>= 1) ss += __shfl_xor(ss, off, 32);
  if ((tid & 31) == 0) red[tid >> 5] = ss;
  __syncthreads();
  if (tid == 0) {
    float t = 0.f;
    #pragma unroll
    for (int i = 0; i < 8; ++i) t += red[i];
    s_scale = rsqrtf(t * (1.0f / Dc) + 1.1920929e-07f);
  }
  __syncthreads();
  const float sc = s_scale;
  #pragma unroll
  for (int i = 0; i < 4; ++i)
    out[(size_t)row * Dc + tid + i * 256] = (__bf16)(v[i] * sc * w[tid + i * 256]);
}

// ---------------------------------------------------------------------------
// Forget gate: f[row,h] = sigmoid(xn[row,:] . Wf[h,:]), one wave per row
// ---------------------------------------------------------------------------
__global__ __launch_bounds__(256) void fgate_k(const __bf16* __restrict__ xn,
                                               const float* __restrict__ Wf,
                                               float* __restrict__ f)
{
  const int lane = threadIdx.x & 31;
  const int wave = threadIdx.x >> 5;
  const int row  = blockIdx.x * 8 + wave;
  const __bf16* xr = xn + (size_t)row * Dc;
  float xv[32];
  #pragma unroll
  for (int i = 0; i < 32; ++i) xv[i] = (float)xr[lane + i * 32];
  #pragma unroll
  for (int h = 0; h < Hc; ++h) {
    const float* wr = Wf + h * Dc;
    float s = 0.f;
    #pragma unroll
    for (int i = 0; i < 32; ++i) s += xv[i] * wr[lane + i * 32];
    #pragma unroll
    for (int off = 16; off >= 1; off >>= 1) s += __shfl_xor(s, off, 32);
    if (lane == 0) f[(size_t)row * Hc + h] = 1.f / (1.f + __expf(-s));
  }
}

// ---------------------------------------------------------------------------
// Sequential gated scan: one block per (b,h); lane j owns state column S[:,j]
// (128 fp32 in registers). S_t = f*S + k (x) v ; o_t = q . S_t
// ---------------------------------------------------------------------------
__global__ __launch_bounds__(128) void scan_k(const float* __restrict__ q,
                                              const float* __restrict__ k,
                                              const float* __restrict__ v,
                                              const float* __restrict__ f,
                                              float* __restrict__ o)
{
  const int b = blockIdx.x / Hc;
  const int h = blockIdx.x % Hc;
  const int j = threadIdx.x;
  __shared__ float qs[DKc];
  __shared__ float ks[DKc];
  float S[DKc];
  #pragma unroll
  for (int i = 0; i < DKc; ++i) S[i] = 0.f;
  const float qsc = 0.08838834764831845f;  // DK^-0.5
  for (int t = 0; t < Sc; ++t) {
    const size_t ro = ((size_t)(b * Sc + t)) * (Hc * DVc) + h * DVc;
    __syncthreads();
    qs[j] = q[ro + j] * qsc;
    ks[j] = k[ro + j];
    const float vj = v[ro + j];
    const float ft = f[(size_t)(b * Sc + t) * Hc + h];
    __syncthreads();
    float ot = 0.f;
    #pragma unroll
    for (int i = 0; i < DKc; ++i) {
      S[i] = ft * S[i] + ks[i] * vj;
      ot += qs[i] * S[i];
    }
    o[ro + j] = ot;
  }
}

// ---------------------------------------------------------------------------
// out = bf16( silu(a) * b )
// ---------------------------------------------------------------------------
__global__ void silu_mul_k(const float* __restrict__ a, const float* __restrict__ b,
                           __bf16* __restrict__ out, int n)
{
  int i = blockIdx.x * 256 + threadIdx.x;
  if (i < n) {
    float g = a[i];
    out[i] = (__bf16)((g / (1.f + __expf(-g))) * b[i]);
  }
}

__global__ void cvt_bf16_k(const float* __restrict__ in, __bf16* __restrict__ out, int n)
{
  int i = blockIdx.x * 256 + threadIdx.x;
  if (i < n) out[i] = (__bf16)in[i];
}

// ---------------------------------------------------------------------------
extern "C" void kernel_launch(void* const* d_in, const int* in_sizes, int n_in,
                              void* d_out, int out_size, void* d_ws, size_t ws_size,
                              hipStream_t stream)
{
  const float* x   = (const float*)d_in[0];
  const float* n1w = (const float*)d_in[1];
  const float* Wq  = (const float*)d_in[2];
  const float* Wk  = (const float*)d_in[3];
  const float* Wv  = (const float*)d_in[4];
  const float* Wf  = (const float*)d_in[5];
  const float* Ww  = (const float*)d_in[6];
  const float* Wo  = (const float*)d_in[7];
  const float* n2w = (const float*)d_in[8];
  const float* w1  = (const float*)d_in[9];
  const float* w2  = (const float*)d_in[10];
  const float* w3  = (const float*)d_in[11];

  char* base = (char*)d_ws;
  size_t off = 0;
  auto alloc = [&](size_t bytes) -> char* {
    char* r = base + off;
    off += (bytes + 255) & ~(size_t)255;
    return r;
  };

  const size_t szW  = (size_t)Dc * Dc;    // 1M elems
  const size_t szWf = (size_t)FFc * Dc;   // 2.88M elems

  // persistent: bf16 weights + x1 residual
  __bf16* wq_b = (__bf16*)alloc(szW * 2);
  __bf16* wk_b = (__bf16*)alloc(szW * 2);
  __bf16* wv_b = (__bf16*)alloc(szW * 2);
  __bf16* ww_b = (__bf16*)alloc(szW * 2);
  __bf16* wo_b = (__bf16*)alloc(szW * 2);
  __bf16* w1_b = (__bf16*)alloc(szWf * 2);
  __bf16* w2_b = (__bf16*)alloc(szWf * 2);
  __bf16* w3_b = (__bf16*)alloc(szWf * 2);
  float*  x1   = (float*)alloc((size_t)Mc * Dc * 4);

  // overlapping pool for phase 1 (mixer) and phase 2 (MLP)
  char* pool = base + off;
  size_t poff = 0;
  auto palloc = [&](size_t bytes) -> char* {
    char* r = pool + poff;
    poff += (bytes + 255) & ~(size_t)255;
    return r;
  };

  // phase 1 buffers
  __bf16* xn  = (__bf16*)palloc((size_t)Mc * Dc * 2);
  float*  qb  = (float*)palloc((size_t)Mc * Dc * 4);
  float*  kb  = (float*)palloc((size_t)Mc * Dc * 4);
  float*  vb  = (float*)palloc((size_t)Mc * Dc * 4);
  float*  wgb = (float*)palloc((size_t)Mc * Dc * 4);
  float*  fb  = (float*)palloc((size_t)Mc * Hc * 4);
  float*  ob  = (float*)palloc((size_t)Mc * Dc * 4);
  __bf16* ogb = (__bf16*)palloc((size_t)Mc * Dc * 2);

  const int nW  = (int)szW;
  const int nWf = (int)szWf;
  cvt_bf16_k<<<(nW  + 255) / 256, 256, 0, stream>>>(Wq, wq_b, nW);
  cvt_bf16_k<<<(nW  + 255) / 256, 256, 0, stream>>>(Wk, wk_b, nW);
  cvt_bf16_k<<<(nW  + 255) / 256, 256, 0, stream>>>(Wv, wv_b, nW);
  cvt_bf16_k<<<(nW  + 255) / 256, 256, 0, stream>>>(Ww, ww_b, nW);
  cvt_bf16_k<<<(nW  + 255) / 256, 256, 0, stream>>>(Wo, wo_b, nW);
  cvt_bf16_k<<<(nWf + 255) / 256, 256, 0, stream>>>(w1, w1_b, nWf);
  cvt_bf16_k<<<(nWf + 255) / 256, 256, 0, stream>>>(w2, w2_b, nWf);
  cvt_bf16_k<<<(nWf + 255) / 256, 256, 0, stream>>>(w3, w3_b, nWf);

  rmsnorm_k<<<Mc, 256, 0, stream>>>(x, n1w, xn);

  dim3 gDD(Dc / 128, Mc / 128);       // (8, 32)
  gemm_bf16<0><<<gDD, 256, 0, stream>>>(xn, wq_b, qb,  nullptr, Mc, Dc, Dc);
  gemm_bf16<0><<<gDD, 256, 0, stream>>>(xn, wk_b, kb,  nullptr, Mc, Dc, Dc);
  gemm_bf16<0><<<gDD, 256, 0, stream>>>(xn, wv_b, vb,  nullptr, Mc, Dc, Dc);
  gemm_bf16<0><<<gDD, 256, 0, stream>>>(xn, ww_b, wgb, nullptr, Mc, Dc, Dc);
  fgate_k<<<Mc / 8, 256, 0, stream>>>(xn, Wf, fb);

  scan_k<<<Bc * Hc, 128, 0, stream>>>(qb, kb, vb, fb, ob);

  {
    const int n = Mc * Dc;
    silu_mul_k<<<(n + 255) / 256, 256, 0, stream>>>(wgb, ob, ogb, n);
  }
  gemm_bf16<1><<<gDD, 256, 0, stream>>>(ogb, wo_b, x1, x, Mc, Dc, Dc);

  // phase 2 (reuse pool; stream-ordered so safe)
  poff = 0;
  __bf16* hbn = (__bf16*)palloc((size_t)Mc * Dc * 2);
  float*  g1b = (float*)palloc((size_t)Mc * FFc * 4);
  float*  g2b = (float*)palloc((size_t)Mc * FFc * 4);
  __bf16* hb  = (__bf16*)palloc((size_t)Mc * FFc * 2);

  rmsnorm_k<<<Mc, 256, 0, stream>>>(x1, n2w, hbn);
  dim3 gDF(FFc / 128, Mc / 128);      // (22, 32)
  gemm_bf16<0><<<gDF, 256, 0, stream>>>(hbn, w1_b, g1b, nullptr, Mc, FFc, Dc);
  gemm_bf16<0><<<gDF, 256, 0, stream>>>(hbn, w2_b, g2b, nullptr, Mc, FFc, Dc);
  {
    const int n = Mc * FFc;
    silu_mul_k<<<(n + 255) / 256, 256, 0, stream>>>(g1b, g2b, hb, n);
  }
  gemm_bf16<1><<<gDD, 256, 0, stream>>>(hb, w3_b, (float*)d_out, x1, Mc, Dc, FFc);

  (void)in_sizes; (void)n_in; (void)out_size; (void)ws_size;
}
